// MaskedMultiHeadAttention_74887049773700
// MI455X (gfx1250) — matrix operand order repaired and checked by reading, not compiled, not run on hardware
//
#include <hip/hip_runtime.h>

typedef __attribute__((ext_vector_type(16))) _Float16 v16h;
typedef __attribute__((ext_vector_type(8)))  _Float16 v8h;
typedef __attribute__((ext_vector_type(8)))  float    v8f;

#define NB 4
#define NS 2048
#define ND 1024
#define NH 16
#define NHD 64
#define NM (NB * NS)   // 8192 tokens

// ---------------------------------------------------------------------------
// fp32 -> f16 elementwise
// ---------------------------------------------------------------------------
__global__ void cvt_f32_to_f16(const float* __restrict__ src,
                               _Float16* __restrict__ dst, int n) {
  int i = blockIdx.x * blockDim.x + threadIdx.x;
  if (i < n) dst[i] = (_Float16)src[i];
}

// W [d][d] row-major fp32 -> Wt f16 with Wt[n*d + k] = W[k*d + n] (col-major)
__global__ void cvt_transpose_f16(const float* __restrict__ W,
                                  _Float16* __restrict__ Wt, int d) {
  int i = blockIdx.x * blockDim.x + threadIdx.x;
  if (i < d * d) {
    int n = i / d;
    int k = i - n * d;
    Wt[(size_t)n * d + k] = (_Float16)W[(size_t)k * d + n];
  }
}

// ---------------------------------------------------------------------------
// WMMA GEMM:  C[M=8192, N=1024] = A_f16(row-major) @ Bt_f16(col-major)
// block = 128 threads (4 waves); wave computes a 32x64 tile, K-steps of 32.
// Per K-step: 2 A operand loads + 4 B operand loads feed 8 WMMAs.
// mode 0: store f16 row-major                          (Q, K)
// mode 1: store f16 per-head transposed [b][h][hd][s]  (V)
// mode 2: store f32 row-major + bias                   (final projection)
// ---------------------------------------------------------------------------
__global__ void gemm_wmma(const _Float16* __restrict__ A,
                          const _Float16* __restrict__ Bt,
                          void* __restrict__ dst,
                          const float* __restrict__ bias, int mode) {
  const int lane = threadIdx.x & 31;
  const int wave = threadIdx.x >> 5;
  const int m0 = blockIdx.x * 32;
  const int n0 = blockIdx.y * 256 + wave * 64;

  v8f acc0[4] = {};  // rows m0 +  0..15
  v8f acc1[4] = {};  // rows m0 + 16..31

  const _Float16* Arow0 = A + (size_t)(m0 + (lane & 15)) * ND;
  const _Float16* Arow1 = Arow0 + (size_t)16 * ND;
  const int kb8  = (lane >> 4) * 16;  // A K-chunk base within step
  const int kb16 = (lane >> 4) * 16;  // B K-chunk base within step

  for (int kk = 0; kk < ND; kk += 32) {
    v8h l0 = *(const v8h*)(Arow0 + kk + kb8);
    v8h h0 = *(const v8h*)(Arow0 + kk + kb8 + 8);
    v8h l1 = *(const v8h*)(Arow1 + kk + kb8);
    v8h h1 = *(const v8h*)(Arow1 + kk + kb8 + 8);
    v16h a0, a1;
#pragma unroll
    for (int i = 0; i < 8; ++i) {
      a0[i] = l0[i]; a0[i + 8] = h0[i];
      a1[i] = l1[i]; a1[i + 8] = h1[i];
    }

#pragma unroll
    for (int t = 0; t < 4; ++t) {
      const int n = n0 + t * 16 + (lane & 15);
      v16h b = *(const v16h*)(Bt + (size_t)n * ND + kk + kb16);
      acc0[t] = __builtin_amdgcn_wmma_f32_16x16x32_f16(
          false, a0, false, b, (short)0, acc0[t], false, false);
      acc1[t] = __builtin_amdgcn_wmma_f32_16x16x32_f16(
          false, a1, false, b, (short)0, acc1[t], false, false);
    }
  }

  const int rbase = (lane >> 4) << 3;  // C-layout: lanes>=16 hold rows 8..15
  const int cw = lane & 15;
#pragma unroll
  for (int g = 0; g < 2; ++g) {
#pragma unroll
    for (int t = 0; t < 4; ++t) {
#pragma unroll
      for (int r = 0; r < 8; ++r) {
        const int row = m0 + g * 16 + rbase + r;
        const int n = n0 + t * 16 + cw;
        const float v = g ? acc1[t][r] : acc0[t][r];
        if (mode == 0) {
          ((_Float16*)dst)[(size_t)row * ND + n] = (_Float16)v;
        } else if (mode == 1) {
          const int bI = row >> 11;          // token / S
          const int sI = row & (NS - 1);
          const int h  = n >> 6;
          const int hd = n & (NHD - 1);
          ((_Float16*)dst)[(((size_t)(bI * NH + h) * NHD + hd) << 11) + sI] =
              (_Float16)v;
        } else {
          ((float*)dst)[(size_t)row * ND + n] = v + bias[n];
        }
      }
    }
  }
}

// ---------------------------------------------------------------------------
// Causal flash attention. 1 wave / block; block handles 16 queries of (b,h),
// key steps of 64 (4 score tiles + 8 PV WMMAs per step; shuffle trees for the
// row max/sum are amortized across all 64 keys of the step).
// Q,K: f16 [B*S][D] row-major.  Vt: f16 [b][h][hd][s].  ctx out: f16 [B*S][D].
// softmax scale = 1/HD^2 = 1/4096 (matches reference).
// ---------------------------------------------------------------------------
__global__ void attn_wmma(const _Float16* __restrict__ Q,
                          const _Float16* __restrict__ K,
                          const _Float16* __restrict__ Vt,
                          _Float16* __restrict__ ctx) {
  __shared__ __align__(16) _Float16 Pb[16 * 64];  // P tile: 16 q x 64 keys

  const int lane = threadIdx.x & 31;
  const int qt = blockIdx.x & 127;       // S/16 = 128 query tiles
  const int bh = blockIdx.x >> 7;
  const int b = bh >> 4;
  const int h = bh & 15;
  const int q0 = qt * 16;

  const int kb8  = (lane >> 4) * 16;
  const int kb16 = (lane >> 4) * 16;
  const int cw = lane & 15;
  const int rbase = (lane >> 4) << 3;

  // Q tile as two A-operands (hd 0..31, hd 32..63)
  const _Float16* Qrow = Q + (size_t)(b * NS + q0 + (lane & 15)) * ND + h * NHD;
  v16h aQ0, aQ1;
  {
    v8h l0 = *(const v8h*)(Qrow + kb8);
    v8h h0 = *(const v8h*)(Qrow + kb8 + 8);
    v8h l1 = *(const v8h*)(Qrow + 32 + kb8);
    v8h h1 = *(const v8h*)(Qrow + 32 + kb8 + 8);
#pragma unroll
    for (int i = 0; i < 8; ++i) {
      aQ0[i] = l0[i]; aQ0[i + 8] = h0[i];
      aQ1[i] = l1[i]; aQ1[i + 8] = h1[i];
    }
  }

  v8f acc[4] = {};
  float mrow[8], lrow[8];
#pragma unroll
  for (int r = 0; r < 8; ++r) { mrow[r] = -1e30f; lrow[r] = 0.0f; }

  const _Float16* Vb = Vt + (size_t)(b * NH + h) * NHD * NS;
  const float inv_scale = 1.0f / 4096.0f;
  const int ntiles = (q0 + 15) / 64 + 1;

  for (int jt = 0; jt < ntiles; ++jt) {
    const int j0 = jt * 64;

    // ---- scores: four 16x16 f32 tiles over 64 keys ------------------------
    v8f s[4] = {};
#pragma unroll
    for (int c = 0; c < 4; ++c) {
      const _Float16* Kr =
          K + (size_t)(b * NS + j0 + c * 16 + cw) * ND + h * NHD;
      v16h bk;
      bk = *(const v16h*)(Kr + kb16);
      s[c] = __builtin_amdgcn_wmma_f32_16x16x32_f16(false, aQ0, false, bk,
                                                    (short)0, s[c], false,
                                                    false);
      bk = *(const v16h*)(Kr + 32 + kb16);
      s[c] = __builtin_amdgcn_wmma_f32_16x16x32_f16(false, aQ1, false, bk,
                                                    (short)0, s[c], false,
                                                    false);
    }

    // ---- scale + causal mask + online softmax -----------------------------
#pragma unroll
    for (int r = 0; r < 8; ++r) {
      const int q = q0 + rbase + r;
      float a[4];
#pragma unroll
      for (int c = 0; c < 4; ++c) {
        float v = s[c][r] * inv_scale;
        a[c] = ((j0 + c * 16 + cw) <= q) ? v : -1e30f;
      }

      float mx = fmaxf(fmaxf(a[0], a[1]), fmaxf(a[2], a[3]));
#pragma unroll
      for (int off = 1; off < 16; off <<= 1)
        mx = fmaxf(mx, __shfl_xor(mx, off, 32));
      const float mnew = fmaxf(mrow[r], mx);
      const float sc = __expf(mrow[r] - mnew);
      mrow[r] = mnew;

      float rs = 0.0f;
#pragma unroll
      for (int c = 0; c < 4; ++c) {
        a[c] = __expf(a[c] - mnew);
        rs += a[c];
      }
#pragma unroll
      for (int off = 1; off < 16; off <<= 1) rs += __shfl_xor(rs, off, 32);
      lrow[r] = lrow[r] * sc + rs;
#pragma unroll
      for (int t = 0; t < 4; ++t) acc[t][r] *= sc;
#pragma unroll
      for (int c = 0; c < 4; ++c) s[c][r] = a[c];
    }

    // ---- stage P through LDS to re-layout C-layout -> A-operand -----------
#pragma unroll
    for (int r = 0; r < 8; ++r) {
#pragma unroll
      for (int c = 0; c < 4; ++c)
        Pb[(rbase + r) * 64 + c * 16 + cw] = (_Float16)s[c][r];
    }
    __syncthreads();
    v16h aP0, aP1;
    {
      const int m = lane & 15;
      v8h plo0 = *(const v8h*)&Pb[m * 64 + kb8];
      v8h phi0 = *(const v8h*)&Pb[m * 64 + kb8 + 8];
      v8h plo1 = *(const v8h*)&Pb[m * 64 + 32 + kb8];
      v8h phi1 = *(const v8h*)&Pb[m * 64 + 32 + kb8 + 8];
#pragma unroll
      for (int i = 0; i < 8; ++i) {
        aP0[i] = plo0[i]; aP0[i + 8] = phi0[i];
        aP1[i] = plo1[i]; aP1[i + 8] = phi1[i];
      }
    }
    __syncthreads();

    // ---- acc += P @ V : two K=32 key chunks, four 16-col hd tiles ---------
#pragma unroll
    for (int t = 0; t < 4; ++t) {
      const int hd = t * 16 + cw;
      v16h bv0 = *(const v16h*)(Vb + (size_t)hd * NS + j0 + kb16);
      v16h bv1 = *(const v16h*)(Vb + (size_t)hd * NS + j0 + 32 + kb16);
      acc[t] = __builtin_amdgcn_wmma_f32_16x16x32_f16(
          false, aP0, false, bv0, (short)0, acc[t], false, false);
      acc[t] = __builtin_amdgcn_wmma_f32_16x16x32_f16(
          false, aP1, false, bv1, (short)0, acc[t], false, false);
    }
  }

  // ---- normalize + store ctx (f16 row-major) ------------------------------
#pragma unroll
  for (int t = 0; t < 4; ++t) {
#pragma unroll
    for (int r = 0; r < 8; ++r) {
      const int token = b * NS + q0 + rbase + r;
      const float o = acc[t][r] / lrow[r];
      ctx[(size_t)token * ND + h * NHD + t * 16 + cw] = (_Float16)o;
    }
  }
}

// ---------------------------------------------------------------------------
extern "C" void kernel_launch(void* const* d_in, const int* in_sizes, int n_in,
                              void* d_out, int out_size, void* d_ws,
                              size_t ws_size, hipStream_t stream) {
  const float* x  = (const float*)d_in[0];
  const float* Wq = (const float*)d_in[1];
  const float* Wk = (const float*)d_in[2];
  const float* Wv = (const float*)d_in[3];
  const float* Wo = (const float*)d_in[4];
  const float* bo = (const float*)d_in[5];

  char* ws = (char*)d_ws;
  const size_t X16_SZ = (size_t)NM * ND * 2;       // 16 MB
  const size_t W_SZ   = (size_t)ND * ND * 2;       // 2 MB each
  _Float16* X16 = (_Float16*)(ws);
  _Float16* Wqt = (_Float16*)(ws + X16_SZ);
  _Float16* Wkt = (_Float16*)(ws + X16_SZ + W_SZ);
  _Float16* Wvt = (_Float16*)(ws + X16_SZ + 2 * W_SZ);
  _Float16* Wot = (_Float16*)(ws + X16_SZ + 3 * W_SZ);
  _Float16* Q16 = (_Float16*)(ws + X16_SZ + 4 * W_SZ);
  _Float16* K16 = (_Float16*)(ws + 2 * X16_SZ + 4 * W_SZ);
  _Float16* Vt16 = (_Float16*)(ws + 3 * X16_SZ + 4 * W_SZ);
  _Float16* CTX = (_Float16*)(ws + 4 * X16_SZ + 4 * W_SZ);

  const int nx = NM * ND;  // 8388608
  hipLaunchKernelGGL(cvt_f32_to_f16, dim3((nx + 255) / 256), dim3(256), 0,
                     stream, x, X16, nx);
  const int nw = ND * ND;
  hipLaunchKernelGGL(cvt_transpose_f16, dim3((nw + 255) / 256), dim3(256), 0,
                     stream, Wq, Wqt, ND);
  hipLaunchKernelGGL(cvt_transpose_f16, dim3((nw + 255) / 256), dim3(256), 0,
                     stream, Wk, Wkt, ND);
  hipLaunchKernelGGL(cvt_transpose_f16, dim3((nw + 255) / 256), dim3(256), 0,
                     stream, Wv, Wvt, ND);
  hipLaunchKernelGGL(cvt_transpose_f16, dim3((nw + 255) / 256), dim3(256), 0,
                     stream, Wo, Wot, ND);

  const dim3 ggrid(NM / 32, ND / 256);  // (256, 4)
  hipLaunchKernelGGL(gemm_wmma, ggrid, dim3(128), 0, stream, X16, Wqt,
                     (void*)Q16, (const float*)nullptr, 0);
  hipLaunchKernelGGL(gemm_wmma, ggrid, dim3(128), 0, stream, X16, Wkt,
                     (void*)K16, (const float*)nullptr, 0);
  hipLaunchKernelGGL(gemm_wmma, ggrid, dim3(128), 0, stream, X16, Wvt,
                     (void*)Vt16, (const float*)nullptr, 1);

  hipLaunchKernelGGL(attn_wmma, dim3(NB * NH * (NS / 16)), dim3(32), 0, stream,
                     Q16, K16, Vt16, CTX);

  hipLaunchKernelGGL(gemm_wmma, ggrid, dim3(128), 0, stream, CTX, Wot, d_out,
                     bo, 2);
}
